// AttentionalFactorizationMachineLayer_15418932593375
// MI455X (gfx1250) — matrix operand order repaired and checked by reading, not compiled, not run on hardware
//
#include <hip/hip_runtime.h>

typedef __attribute__((ext_vector_type(16))) _Float16 v16h;
typedef __attribute__((ext_vector_type(8)))  _Float16 v8h;
typedef __attribute__((ext_vector_type(8)))  float    v8f;

#define NF      32
#define EMB     64
#define NPAIR   496
#define HSTRIDE 72          // padded f16 row stride (72 halves = 144B) -> conflict-free b128 LDS reads

// ---- dynamic LDS layout (bytes), all 16B aligned ----
#define OFF_XS      0        // 2048 f32            (x tile, f32, kept for exact final reduction)
#define OFF_INNER   8192     // 496*72 f16          (pairwise products, WMMA A staging; reused as scratch later)
#define OFF_W1T     79616    // 64*72 f16           (W1 transposed, WMMA B staging)
#define OFF_B1      88832    // 64 f32
#define OFF_W2      89088    // 64 f32
#define OFF_LOGITS  89344    // 496 f32
#define OFF_ATTN    91328    // 496 f32
#define OFF_RED     93312    // 256 f32
#define OFF_PR      94336    // 496 short
#define OFF_PC      95328    // 496 short
#define SMEM_BYTES  96320

// VALU-only XOR-butterfly add within each 16-lane row via v_permlane16_b32.
// Selector nibbles: dst lane i reads src lane sel[i] (row-local), sel[i] = i ^ mask.
__device__ __forceinline__ float permlane16_xor_add(float v, unsigned s0, unsigned s1) {
    int u = __builtin_bit_cast(int, v);
    int p = __builtin_amdgcn_permlane16(u, u, (int)s0, (int)s1, false, false);
    return v + __builtin_bit_cast(float, p);
}

__global__ void __launch_bounds__(256)
afm_fused_kernel(const float* __restrict__ x,    // (B,32,64)
                 const float* __restrict__ W1,   // (64,64)
                 const float* __restrict__ b1,   // (64)
                 const float* __restrict__ w2,   // (64,1)
                 float* __restrict__ outVec,     // (B,64)
                 float* __restrict__ attnOut)    // (B,496)
{
    extern __shared__ unsigned char smem[];
    float*    xs      = (float*)   (smem + OFF_XS);
    _Float16* innerH  = (_Float16*)(smem + OFF_INNER);
    _Float16* w1t     = (_Float16*)(smem + OFF_W1T);
    float*    b1s     = (float*)   (smem + OFF_B1);
    float*    w2s     = (float*)   (smem + OFF_W2);
    float*    logitsS = (float*)   (smem + OFF_LOGITS);
    float*    attnS   = (float*)   (smem + OFF_ATTN);
    float*    red     = (float*)   (smem + OFF_RED);
    short*    prS     = (short*)   (smem + OFF_PR);
    short*    pcS     = (short*)   (smem + OFF_PC);

    const int tid  = threadIdx.x;
    const int bIdx = blockIdx.x;
    const int lane = tid & 31;
    const int nlo  = lane & 15;     // column / row-in-group selector
    const int grp  = lane >> 4;     // lane half-group
    const int wave = tid >> 5;

    // ---- triu_indices(32, k=1) pair table ----
    for (int p = tid; p < NPAIR; p += 256) {
        int i = 0, base = 0;
        while (base + (NF - 1 - i) <= p) { base += NF - 1 - i; ++i; }
        prS[p] = (short)i;
        pcS[p] = (short)(i + 1 + (p - base));
    }

    // ---- stage x[b] (f32, float4 vectorized), W1^T (f16, padded), b1, w2 ----
    const float4* xb4 = (const float4*)(x + (size_t)bIdx * (NF * EMB));
    float4*       xs4 = (float4*)xs;
#pragma unroll
    for (int idx = tid; idx < (NF * EMB) / 4; idx += 256) xs4[idx] = xb4[idx];
    for (int idx = tid; idx < EMB * EMB; idx += 256) {
        int e = idx >> 6, a = idx & 63;
        w1t[a * HSTRIDE + e] = (_Float16)W1[idx];
    }
    if (tid < 64) { b1s[tid] = b1[tid]; w2s[tid] = w2[tid]; }
    __syncthreads();

    // ---- inner[p][e] = x[row][e]*x[col][e] -> f16 (WMMA A operand), 8 elems/unit ----
    for (int u = tid; u < NPAIR * 8; u += 256) {
        const int p  = u >> 3;
        const int e0 = (u & 7) * 8;
        const float* ra = &xs[prS[p] * EMB + e0];
        const float* rb = &xs[pcS[p] * EMB + e0];
        float4 a0 = *(const float4*)(ra);
        float4 a1 = *(const float4*)(ra + 4);
        float4 c0 = *(const float4*)(rb);
        float4 c1 = *(const float4*)(rb + 4);
        v8h h;
        h[0] = (_Float16)(a0.x * c0.x);
        h[1] = (_Float16)(a0.y * c0.y);
        h[2] = (_Float16)(a0.z * c0.z);
        h[3] = (_Float16)(a0.w * c0.w);
        h[4] = (_Float16)(a1.x * c1.x);
        h[5] = (_Float16)(a1.y * c1.y);
        h[6] = (_Float16)(a1.z * c1.z);
        h[7] = (_Float16)(a1.w * c1.w);
        *(v8h*)(&innerH[p * HSTRIDE + e0]) = h;
    }
    __syncthreads();

    // ---- preload B fragments: W1^T, 4 N-tiles x 2 K-blocks ----
    // B 32x16 f16 layout: lane holds col n = nlo; lanes 0-15 K=0..15, lanes 16-31 K=16..31
    v16h bf[4][2];
#pragma unroll
    for (int nb = 0; nb < 4; ++nb) {
        int n = nb * 16 + nlo;
#pragma unroll
        for (int kb = 0; kb < 2; ++kb) {
            const _Float16* src = &w1t[n * HSTRIDE + kb * 32 + grp * 16];
            v8h lo = *(const v8h*)(src);
            v8h hi = *(const v8h*)(src + 8);
            bf[nb][kb] = __builtin_shufflevector(lo, hi, 0,1,2,3,4,5,6,7,8,9,10,11,12,13,14,15);
        }
    }

    // ---- per-wave M-tiles: GEMM (WMMA) + bias + ReLU + dot(w2) -> logits ----
    for (int t = wave; t < 31; t += 8) {
        const int p0 = t * 16;
        // A 16x32 f16 layout: lane holds row p0+nlo; grp0 K=0..15, grp1 K=16..31
        v16h af[2];
#pragma unroll
        for (int kb = 0; kb < 2; ++kb) {
            const _Float16* src = &innerH[(p0 + nlo) * HSTRIDE + kb * 32 + grp * 16];
            v8h lo = *(const v8h*)(src);
            v8h hi = *(const v8h*)(src + 8);
            af[kb] = __builtin_shufflevector(lo, hi, 0,1,2,3,4,5,6,7,8,9,10,11,12,13,14,15);
        }
        float acc[8] = {0.f,0.f,0.f,0.f,0.f,0.f,0.f,0.f};
#pragma unroll
        for (int nb = 0; nb < 4; ++nb) {
            v8f c = {};
            c = __builtin_amdgcn_wmma_f32_16x16x32_f16(false, af[0], false, bf[nb][0],
                                                       (short)0, c, false, false);
            c = __builtin_amdgcn_wmma_f32_16x16x32_f16(false, af[1], false, bf[nb][1],
                                                       (short)0, c, false, false);
            const int a  = nb * 16 + nlo;
            const float bb = b1s[a], ww = w2s[a];
#pragma unroll
            for (int r = 0; r < 8; ++r) {           // D: VGPR r -> row p0 + r + 8*grp, col a
                float hv = fmaxf(c[r] + bb, 0.0f);  // +b1, ReLU
                acc[r] += hv * ww;                  // partial logit over this lane's columns
            }
        }
        // reduce acc[r] across the 16 columns of the lane group (pure VALU butterflies)
#pragma unroll
        for (int r = 0; r < 8; ++r) {
            float v = acc[r];
            v = permlane16_xor_add(v, 0x67452301u, 0xEFCDAB89u);  // xor 1
            v = permlane16_xor_add(v, 0x54761032u, 0xDCFE98BAu);  // xor 2
            v = permlane16_xor_add(v, 0x32107654u, 0xBA98FEDCu);  // xor 4
            v = permlane16_xor_add(v, 0xFEDCBA98u, 0x76543210u);  // xor 8
            acc[r] = v;
        }
#pragma unroll
        for (int r = 0; r < 8; ++r)
            if (nlo == 0) logitsS[p0 + r + 8 * grp] = acc[r];
    }
    __syncthreads();

    // ---- softmax over 496 logits (b2 is softmax-invariant -> dropped) ----
    float v0 = (tid < NPAIR)       ? logitsS[tid]       : -3.0e38f;
    float v1 = (tid + 256 < NPAIR) ? logitsS[tid + 256] : -3.0e38f;
    red[tid] = fmaxf(v0, v1);
    __syncthreads();
    for (int s = 128; s > 0; s >>= 1) {
        if (tid < s) red[tid] = fmaxf(red[tid], red[tid + s]);
        __syncthreads();
    }
    const float mx = red[0];
    __syncthreads();
    float e0 = (tid < NPAIR)       ? __expf(v0 - mx) : 0.0f;
    float e1 = (tid + 256 < NPAIR) ? __expf(v1 - mx) : 0.0f;
    red[tid] = e0 + e1;
    __syncthreads();
    for (int s = 128; s > 0; s >>= 1) {
        if (tid < s) red[tid] += red[tid + s];
        __syncthreads();
    }
    const float inv = 1.0f / red[0];
    __syncthreads();
    float* aOut = attnOut + (size_t)bIdx * NPAIR;
    if (tid < NPAIR)       { float a0 = e0 * inv; attnS[tid]       = a0; aOut[tid]       = a0; }
    if (tid + 256 < NPAIR) { float a1 = e1 * inv; attnS[tid + 256] = a1; aOut[tid + 256] = a1; }
    __syncthreads();

    // ---- out[e] = sum_p attn[p]*x[r][e]*x[c][e]  (f32, float4 over embed dim) ----
    // 16 threads * float4 cover e=0..63; 16 chunks of 31 pairs. innerH is dead -> reuse as scratch.
    float* wsum = (float*)(smem + OFF_INNER);      // 16 chunks * 64 floats = 4KB
    {
        const int e4    = (tid & 15) * 4;
        const int chunk = tid >> 4;
        float4 part = make_float4(0.f, 0.f, 0.f, 0.f);
        for (int p = chunk; p < NPAIR; p += 16) {
            const float a = attnS[p];
            const float4 xr = *(const float4*)(&xs[prS[p] * EMB + e4]);
            const float4 xc = *(const float4*)(&xs[pcS[p] * EMB + e4]);
            part.x += a * xr.x * xc.x;
            part.y += a * xr.y * xc.y;
            part.z += a * xr.z * xc.z;
            part.w += a * xr.w * xc.w;
        }
        *(float4*)(&wsum[chunk * EMB + e4]) = part;
    }
    __syncthreads();
    if (tid < EMB) {
        float s = 0.f;
#pragma unroll
        for (int c = 0; c < 16; ++c) s += wsum[c * EMB + tid];
        outVec[(size_t)bIdx * EMB + tid] = s;
    }
}

extern "C" void kernel_launch(void* const* d_in, const int* in_sizes, int n_in,
                              void* d_out, int out_size, void* d_ws, size_t ws_size,
                              hipStream_t stream) {
    const float* x  = (const float*)d_in[0];
    const float* W1 = (const float*)d_in[1];
    const float* b1 = (const float*)d_in[2];
    const float* w2 = (const float*)d_in[3];
    // d_in[4] = b2 : softmax-invariant, unused.
    const int B = in_sizes[0] / (NF * EMB);   // 2048
    float* out  = (float*)d_out;              // (B,1,64) flat
    float* attn = out + (size_t)B * EMB;      // (B,496,1) flat
    hipLaunchKernelGGL(afm_fused_kernel, dim3(B), dim3(256), SMEM_BYTES, stream,
                       x, W1, b1, w2, out, attn);
}
